// RoussAttentionBlock_75376676045506
// MI455X (gfx1250) — matrix operand order repaired and checked by reading, not compiled, not run on hardware
//
#include <hip/hip_runtime.h>

// ---------------------------------------------------------------- constants
constexpr int kB     = 16;
constexpr int kC     = 512;
constexpr int kL     = 1024;
constexpr int kHeads = 8;
constexpr int kCh    = 64;     // channels per head
constexpr int kO3    = 1536;   // 3*c
constexpr float kScale = 0.35355339059327379f; // 1/sqrt(sqrt(64))

typedef __attribute__((ext_vector_type(16))) _Float16 v16h;
typedef __attribute__((ext_vector_type(8)))  _Float16 v8h;
typedef __attribute__((ext_vector_type(8)))  float    v8f;

// ---------------------------------------------------------------- fragments
// A fragment: 16x32 f16, source row-major A[m][k], row stride lda (halves).
// Contiguous k order used here (same for A and B): lane m=lane&15, half=lane>>4; VGPR j: k=half*16+2j+{0,1}.
// One contiguous 32B run per lane.
__device__ __forceinline__ v16h load_a_frag(const _Float16* A, int lda, int lane) {
  const int m  = lane & 15;
  const int hf = (lane >> 4) & 1;
  const _Float16* row = A + (size_t)m * lda + hf * 16;
  v16h a;
#pragma unroll
  for (int i = 0; i < 16; ++i) a[i] = row[i];
  return a;
}

// B fragment: 32x16 f16, B[k][n] supplied as BT[n][k] row-major (ldb halves).
// Contiguous k order used here (same for A and B): lane n=lane&15, half=lane>>4; VGPR j: k=half*16+2j+{0,1}.
// One contiguous 32B run per lane.
__device__ __forceinline__ v16h load_bt_frag(const _Float16* BT, int ldb, int lane) {
  const int n  = lane & 15;
  const int hf = (lane >> 4) & 1;
  const _Float16* row = BT + (size_t)n * ldb + hf * 16;
  v16h b;
#pragma unroll
  for (int i = 0; i < 16; ++i) b[i] = row[i];
  return b;
}

__device__ __forceinline__ v8f wmma_f16(v16h a, v16h b, v8f c) {
  // D = A(16x32) * B(32x16) + C, f32 accumulate
  return __builtin_amdgcn_wmma_f32_16x16x32_f16(false, a, false, b, (short)0, c,
                                                false, false);
}

// ---------------------------------------------------------------- kernel 1: GN stats
// One workgroup per (b, group): reduce 16 ch * 1024 = 16384 contiguous floats.
__global__ __launch_bounds__(256) void gn_stats_kernel(const float* __restrict__ x,
                                                       float* __restrict__ stats) {
  const int grp = blockIdx.x;              // 0..511 == b*32+g
  const float* p = x + (size_t)grp * 16384;
  float s = 0.f, ss = 0.f;
  for (int i = threadIdx.x; i < 16384; i += 256) {
    float v = p[i];
    s += v; ss += v * v;
  }
  __shared__ float sb[256], qb[256];
  sb[threadIdx.x] = s; qb[threadIdx.x] = ss;
  __syncthreads();
  for (int off = 128; off > 0; off >>= 1) {
    if ((int)threadIdx.x < off) {
      sb[threadIdx.x] += sb[threadIdx.x + off];
      qb[threadIdx.x] += qb[threadIdx.x + off];
    }
    __syncthreads();
  }
  if (threadIdx.x == 0) {
    float mean = sb[0] * (1.f / 16384.f);
    float var  = qb[0] * (1.f / 16384.f) - mean * mean;
    stats[grp * 2 + 0] = mean;
    stats[grp * 2 + 1] = rsqrtf(var + 1e-5f);
  }
}

// ---------------------------------------------------------------- kernel 2: conv_w -> f16
__global__ void convert_w_kernel(const float* __restrict__ w, _Float16* __restrict__ wf) {
  int i = blockIdx.x * 256 + threadIdx.x;   // grid sized exactly 1536*512/256
  wf[i] = (_Float16)w[i];
}

// ---------------------------------------------------------------- kernel 3: GN apply + transpose
// x[b][c][l] -> hT[b][l][c] (f16), via 32x32 LDS tile so both sides coalesce.
__global__ __launch_bounds__(256) void gn_apply_kernel(const float* __restrict__ x,
                                                       const float* __restrict__ stats,
                                                       const float* __restrict__ gnw,
                                                       const float* __restrict__ gnb,
                                                       _Float16* __restrict__ hT) {
  __shared__ float tile[32][33];
  const int tx = threadIdx.x & 31;
  const int ty = threadIdx.x >> 5;
  const int l0 = blockIdx.x * 32;
  const int c0 = blockIdx.y * 32;
  const int b  = blockIdx.z;
#pragma unroll
  for (int j = 0; j < 4; ++j) {
    const int c = c0 + ty + j * 8;
    const float mean = stats[(b * 32 + (c >> 4)) * 2 + 0];
    const float rstd = stats[(b * 32 + (c >> 4)) * 2 + 1];
    const float v = x[((size_t)b * kC + c) * kL + l0 + tx];
    tile[ty + j * 8][tx] = (v - mean) * rstd * gnw[c] + gnb[c];
  }
  __syncthreads();
#pragma unroll
  for (int j = 0; j < 4; ++j) {
    const int l = l0 + ty + j * 8;
    hT[((size_t)b * kL + l) * kC + c0 + tx] = (_Float16)tile[tx][ty + j * 8];
  }
}

// ---------------------------------------------------------------- kernel 4: QKV GEMM (WMMA)
// qkv[o][l] = sum_c wf[o][c] * hT[l][c], per batch. Each wave: 16(o) x 64(l)
// tile, K=512 in 16 steps of 32, software-pipelined (double-buffered
// fragments) so next step's global_load_b128s overlap current WMMAs.
__global__ __launch_bounds__(256) void qkv_gemm_kernel(const _Float16* __restrict__ wf,
                                                       const _Float16* __restrict__ hT,
                                                       const float* __restrict__ cb,
                                                       _Float16* __restrict__ qT,
                                                       _Float16* __restrict__ kT,
                                                       _Float16* __restrict__ vv) {
  const int lane = threadIdx.x & 31;
  const int wv   = threadIdx.x >> 5;                 // 8 waves / WG
  const int o0   = blockIdx.y * 128 + wv * 16;       // output-channel tile
  const int l0   = blockIdx.x * 64;                  // spatial tile
  const int bb   = blockIdx.z;
  const _Float16* hTb = hT + (size_t)bb * kL * kC;
  const _Float16* Ab  = wf + (size_t)o0 * kC;

  v8f acc[4] = {v8f{}, v8f{}, v8f{}, v8f{}};

  // prologue: fragments for k0 = 0
  v16h a  = load_a_frag(Ab, kC, lane);
  v16h b0 = load_bt_frag(hTb + (size_t)(l0 +  0) * kC, kC, lane);
  v16h b1 = load_bt_frag(hTb + (size_t)(l0 + 16) * kC, kC, lane);
  v16h b2 = load_bt_frag(hTb + (size_t)(l0 + 32) * kC, kC, lane);
  v16h b3 = load_bt_frag(hTb + (size_t)(l0 + 48) * kC, kC, lane);

  for (int k0 = 0; k0 < kC; k0 += 32) {
    const int kn = (k0 + 32 < kC) ? (k0 + 32) : k0;   // next step (dup on last)
    // issue next step's loads first -> they overlap this step's WMMAs
    v16h an  = load_a_frag(Ab + kn, kC, lane);
    v16h nb0 = load_bt_frag(hTb + (size_t)(l0 +  0) * kC + kn, kC, lane);
    v16h nb1 = load_bt_frag(hTb + (size_t)(l0 + 16) * kC + kn, kC, lane);
    v16h nb2 = load_bt_frag(hTb + (size_t)(l0 + 32) * kC + kn, kC, lane);
    v16h nb3 = load_bt_frag(hTb + (size_t)(l0 + 48) * kC + kn, kC, lane);

    acc[0] = wmma_f16(a, b0, acc[0]);
    acc[1] = wmma_f16(a, b1, acc[1]);
    acc[2] = wmma_f16(a, b2, acc[2]);
    acc[3] = wmma_f16(a, b3, acc[3]);

    a = an; b0 = nb0; b1 = nb1; b2 = nb2; b3 = nb3;
  }

  const int hf = lane >> 4, n = lane & 15;
  float bias[8];
#pragma unroll
  for (int r = 0; r < 8; ++r) bias[r] = cb[o0 + hf * 8 + r];

  const int seg  = o0 >> 9;          // 0=q 1=k 2=v (wave-uniform)
  const int oloc = o0 & 511;
  const int head = oloc >> 6;
  const int ch0  = (oloc & 63) + hf * 8;

  if (seg < 2) {
    _Float16* dst = (seg == 0 ? qT : kT) + (size_t)(bb * kHeads + head) * kL * kCh;
#pragma unroll
    for (int nt = 0; nt < 4; ++nt) {
      const int l = l0 + nt * 16 + n;
      v8h ph;
#pragma unroll
      for (int r = 0; r < 8; ++r)
        ph[r] = (_Float16)((acc[nt][r] + bias[r]) * kScale);
      *(v8h*)(dst + (size_t)l * kCh + ch0) = ph;      // 16B store, t-major [t][ch]
    }
  } else {
    _Float16* dst = vv + (size_t)(bb * kHeads + head) * kCh * kL;
#pragma unroll
    for (int nt = 0; nt < 4; ++nt) {
      const int l = l0 + nt * 16 + n;
#pragma unroll
      for (int r = 0; r < 8; ++r)
        dst[(size_t)(ch0 + r) * kL + l] = (_Float16)(acc[nt][r] + bias[r]);
    }
  }
}

// ---------------------------------------------------------------- kernel 5: flash attention + residual
// Wave = one 16-query tile of one (b,head). Stream K/V in s-blocks of 32 with
// online softmax; P reshaped C-frag -> A-frag through per-wave LDS tile.
// Pipelining: V-frag loads issued before the softmax VALU stretch; K frags
// double-buffered so next block's loads overlap softmax + P*V WMMAs.
__global__ __launch_bounds__(128) void attn_kernel(const float* __restrict__ x,
                                                   const _Float16* __restrict__ qT,
                                                   const _Float16* __restrict__ kT,
                                                   const _Float16* __restrict__ vv,
                                                   float* __restrict__ out) {
  const int lane = threadIdx.x & 31;
  const int wv   = threadIdx.x >> 5;                // 4 waves / WG
  const int bh   = blockIdx.y;                      // 0..127
  const int b    = bh >> 3;
  const int t0   = blockIdx.x * 64 + wv * 16;
  const _Float16* q_bh = qT + (size_t)bh * kL * kCh;
  const _Float16* k_bh = kT + (size_t)bh * kL * kCh;
  const _Float16* v_bh = vv + (size_t)bh * kCh * kL;
  const size_t xbase = ((size_t)b * kC + (bh & 7) * kCh) * kL;

  __shared__ __align__(16) _Float16 pbuf[4][16][40];  // per-wave P tile, padded

  const int n = lane & 15, hf = lane >> 4;

  // Q fragments (reused for whole row of s): A = q^T[t][c], two K-steps
  v16h aq0 = load_a_frag(q_bh + (size_t)t0 * kCh, kCh, lane);
  v16h aq1 = load_a_frag(q_bh + (size_t)t0 * kCh + 32, kCh, lane);

  v8f acc[4] = {v8f{}, v8f{}, v8f{}, v8f{}};        // O[16 x 64]
  float mrow[8], lrow[8];
#pragma unroll
  for (int r = 0; r < 8; ++r) { mrow[r] = -1e30f; lrow[r] = 0.f; }

  // prologue: K fragments for s0 = 0
  v16h k00 = load_bt_frag(k_bh + 0 * kCh,       kCh, lane);
  v16h k01 = load_bt_frag(k_bh + 0 * kCh + 32,  kCh, lane);
  v16h k10 = load_bt_frag(k_bh + 16 * kCh,      kCh, lane);
  v16h k11 = load_bt_frag(k_bh + 16 * kCh + 32, kCh, lane);

  for (int s0 = 0; s0 < kL; s0 += 32) {
    // V fragments for this block: issue now, consumed only after softmax,
    // so the exp/shuffle stretch hides their latency.
    v16h bv0 = load_bt_frag(v_bh +  0 * kL + s0, kL, lane);
    v16h bv1 = load_bt_frag(v_bh + 16 * kL + s0, kL, lane);
    v16h bv2 = load_bt_frag(v_bh + 32 * kL + s0, kL, lane);
    v16h bv3 = load_bt_frag(v_bh + 48 * kL + s0, kL, lane);

    // S[16 x 32] = Q(16x64) * K^T(64x32)
    v8f sa = {}, sb = {};
    sa = wmma_f16(aq0, k00, sa);
    sa = wmma_f16(aq1, k01, sa);
    sb = wmma_f16(aq0, k10, sb);
    sb = wmma_f16(aq1, k11, sb);

    // prefetch next block's K fragments (overlap softmax + P*V WMMAs)
    const int sn = (s0 + 32 < kL) ? (s0 + 32) : s0;
    v16h nk00 = load_bt_frag(k_bh + (size_t)sn * kCh,            kCh, lane);
    v16h nk01 = load_bt_frag(k_bh + (size_t)sn * kCh + 32,       kCh, lane);
    v16h nk10 = load_bt_frag(k_bh + (size_t)(sn + 16) * kCh,     kCh, lane);
    v16h nk11 = load_bt_frag(k_bh + (size_t)(sn + 16) * kCh + 32,kCh, lane);

    // Online softmax. Row m=hf*8+r lives in register slot r across a 16-lane
    // half-group -> xor masks 1,2,4,8 are exactly the row reduction.
#pragma unroll
    for (int r = 0; r < 8; ++r) {
      float mx = fmaxf(sa[r], sb[r]);
      mx = fmaxf(mx, __shfl_xor(mx, 1, 32));
      mx = fmaxf(mx, __shfl_xor(mx, 2, 32));
      mx = fmaxf(mx, __shfl_xor(mx, 4, 32));
      mx = fmaxf(mx, __shfl_xor(mx, 8, 32));
      const float nm = fmaxf(mrow[r], mx);
      const float al = __expf(mrow[r] - nm);
      const float p0 = __expf(sa[r] - nm);
      const float p1 = __expf(sb[r] - nm);
      float rs = p0 + p1;
      rs += __shfl_xor(rs, 1, 32);
      rs += __shfl_xor(rs, 2, 32);
      rs += __shfl_xor(rs, 4, 32);
      rs += __shfl_xor(rs, 8, 32);
      lrow[r] = lrow[r] * al + rs;
      mrow[r] = nm;
      acc[0][r] *= al; acc[1][r] *= al; acc[2][r] *= al; acc[3][r] *= al;
      const int m = hf * 8 + r;
      pbuf[wv][m][n]      = (_Float16)p0;   // C-frag -> LDS (per-wave private,
      pbuf[wv][m][16 + n] = (_Float16)p1;   // wave-internal DS ordering)
    }

    // O += P(16x32) * V^T(32x64): A-frag from LDS, B-frags preloaded above
    v16h ap = load_a_frag(&pbuf[wv][0][0], 40, lane);
    acc[0] = wmma_f16(ap, bv0, acc[0]);
    acc[1] = wmma_f16(ap, bv1, acc[1]);
    acc[2] = wmma_f16(ap, bv2, acc[2]);
    acc[3] = wmma_f16(ap, bv3, acc[3]);

    k00 = nk00; k01 = nk01; k10 = nk10; k11 = nk11;
  }

  // out[b][head*64+c][t] = x + O/l ; per lane: fixed c, 8 consecutive t -> 2x float4
#pragma unroll
  for (int nt = 0; nt < 4; ++nt) {
    const size_t base = xbase + (size_t)(nt * 16 + n) * kL + t0 + hf * 8;
#pragma unroll
    for (int q4 = 0; q4 < 2; ++q4) {
      float4 xv = *(const float4*)(x + base + q4 * 4);
      float4 ov;
      ov.x = xv.x + acc[nt][q4 * 4 + 0] / lrow[q4 * 4 + 0];
      ov.y = xv.y + acc[nt][q4 * 4 + 1] / lrow[q4 * 4 + 1];
      ov.z = xv.z + acc[nt][q4 * 4 + 2] / lrow[q4 * 4 + 2];
      ov.w = xv.w + acc[nt][q4 * 4 + 3] / lrow[q4 * 4 + 3];
      *(float4*)(out + base + q4 * 4) = ov;
    }
  }
}

// ---------------------------------------------------------------- launch
extern "C" void kernel_launch(void* const* d_in, const int* in_sizes, int n_in,
                              void* d_out, int out_size, void* d_ws, size_t ws_size,
                              hipStream_t stream) {
  (void)in_sizes; (void)n_in; (void)out_size; (void)ws_size;
  const float* x   = (const float*)d_in[0];
  const float* gnw = (const float*)d_in[1];
  const float* gnb = (const float*)d_in[2];
  const float* cw  = (const float*)d_in[3];
  const float* cb  = (const float*)d_in[4];
  float* out = (float*)d_out;

  char* ws = (char*)d_ws;
  size_t off = 0;
  float*    stats = (float*)(ws + off);    off += 4096;                              // 512*2 f32
  _Float16* wf16  = (_Float16*)(ws + off); off += (size_t)kO3 * kC * 2;              // 1.5 MB
  _Float16* hT    = (_Float16*)(ws + off); off += (size_t)kB * kL * kC * 2;          // 16 MB
  _Float16* qT    = (_Float16*)(ws + off); off += (size_t)kB * kHeads * kL * kCh * 2;
  _Float16* kTp   = (_Float16*)(ws + off); off += (size_t)kB * kHeads * kL * kCh * 2;
  _Float16* vv    = (_Float16*)(ws + off);

  gn_stats_kernel<<<kB * 32, 256, 0, stream>>>(x, stats);
  convert_w_kernel<<<(kO3 * kC) / 256, 256, 0, stream>>>(cw, wf16);
  gn_apply_kernel<<<dim3(kL / 32, kC / 32, kB), 256, 0, stream>>>(x, stats, gnw, gnb, hT);
  qkv_gemm_kernel<<<dim3(kL / 64, kO3 / 128, kB), 256, 0, stream>>>(wf16, hT, cb, qT, kTp, vv);
  attn_kernel<<<dim3(kL / 64, kB * kHeads), 128, 0, stream>>>(x, qT, kTp, vv, out);
}
